// SSMLayer_29875792511770
// MI455X (gfx1250) — compile-verified, matrix operand order checked by reading, not run on hardware
//
#include <hip/hip_runtime.h>
#include <hip/hip_bf16.h>
#include <math.h>

// ---------- types ----------
typedef __attribute__((ext_vector_type(16))) __bf16 v16bf;
typedef __attribute__((ext_vector_type(8)))  __bf16 v8bf;
typedef __attribute__((ext_vector_type(8)))  float  v8f;
typedef __attribute__((ext_vector_type(4)))  unsigned int u32x4;
typedef __attribute__((ext_vector_type(8)))  int i32x8;
typedef __attribute__((ext_vector_type(4)))  int i32x4;

// B=4, S=4096, D=1024, N=16
#define Mrows 16384            // B*S
#define Kdim  1024             // D
#define Ncat  2176             // 1024 (Wd) + 1024 (WD) + 16 (Wb) + 16 (Wc) + 96 pad
#define BM 128
#define BN 128
#define BK 32
#define BKP 40                 // padded LDS row stride in bf16 (80B: conflict-free frag reads)
#define STAGE_USH ((BM + BN) * BKP)       // ushorts per stage
#define STAGE_BYTES (STAGE_USH * 2)       // 20480 B
#define SMEM_BYTES (2 * STAGE_BYTES)      // 40960 B (double buffered)
#define CHUNK 128
#define NCH 32

// ---- Tensor Data Mover availability / arity detection ----
#if defined(__has_builtin)
#  if __has_builtin(__builtin_amdgcn_tensor_load_to_lds) && \
      __has_builtin(__builtin_amdgcn_s_wait_tensorcnt)
#    define USE_TDM 1
#  endif
#endif
#if defined(USE_TDM) && defined(__has_include)
#  if __has_include(<hip/amd_detail/amd_gfx1250_TDM.h>)
#    define TDM_6ARG 1
#  endif
#endif

__device__ __forceinline__ unsigned short f2bf(float f) {
    unsigned u = __float_as_uint(f);
    unsigned r = u + 0x7FFFu + ((u >> 16) & 1u);   // RNE
    return (unsigned short)(r >> 16);
}
__device__ __forceinline__ float bf2f(unsigned short u) {
    return __uint_as_float(((unsigned)u) << 16);
}

#if defined(USE_TDM)
// Issue one TDM 2D tile load: rows x 32 bf16, row stride 1024 elements,
// LDS padded 16B per 64B row (=> 80B LDS row stride).
__device__ __forceinline__ void tdm_load_tile(unsigned lds_byte_off,
                                              const unsigned short* gptr)
{
    unsigned long long ga = (unsigned long long)(uintptr_t)(const void*)gptr;
    u32x4 g0;
    g0.x = 1u;                                     // count=1 (valid), user mode
    g0.y = lds_byte_off;                           // LDS byte address (dynamic LDS base = 0)
    g0.z = (unsigned)(ga & 0xFFFFFFFFu);           // global_addr[31:0]
    g0.w = (unsigned)((ga >> 32) & 0x1FFFFFFu)     // global_addr[56:32]
         | (2u << 30);                             // type = 2 ("image")
    i32x8 g1;
    g1[0] = (int)((1u << 16)        // data_size = 1 -> 2 bytes
                | (1u << 20)        // pad_enable
                | (3u << 22)        // pad_interval: 8B<<3 = 64B
                | (3u << 25));      // pad_amount: (3+1)*4B = 16B
    g1[1] = (int)(32u << 16);       // tensor_dim0[15:0] = 32 (bits 63:48)
    g1[2] = (int)(128u << 16);      // tensor_dim0 hi = 0; tensor_dim1[15:0] = 128
    g1[3] = (int)(32u << 16);       // tensor_dim1 hi = 0; tile_dim0 = 32
    g1[4] = 128;                    // tile_dim1 = 128, tile_dim2 = 0
    g1[5] = 1024;                   // tensor_dim0_stride[31:0] = 1024 elements
    g1[6] = 0;                      // stride hi, dim1_stride lo
    g1[7] = 0;
    i32x4 z4 = {0, 0, 0, 0};
#if defined(TDM_6ARG)
    i32x8 z8 = {0, 0, 0, 0, 0, 0, 0, 0};
    __builtin_amdgcn_tensor_load_to_lds(g0, g1, z4, z4, z8, 0);
#else
    __builtin_amdgcn_tensor_load_to_lds(g0, g1, z4, z4, 0);
#endif
}
#endif // USE_TDM

// ---------------- kernel 0: pack [Wd; WD; Wb; Wc; 0] -> bf16, biases -> bcat ----------------
__global__ __launch_bounds__(256) void pack_kernel(
    const float* __restrict__ Wd, const float* __restrict__ bd,
    const float* __restrict__ Wb, const float* __restrict__ bb,
    const float* __restrict__ Wc, const float* __restrict__ bc,
    const float* __restrict__ WD, const float* __restrict__ bD,
    unsigned short* __restrict__ wcat, float* __restrict__ bcat)
{
    int r = blockIdx.x;                // 0..2175
    int k = threadIdx.x * 4;           // 0..1020
    const float* src = nullptr;
    if      (r < 1024) src = Wd + (size_t)r * Kdim;
    else if (r < 2048) src = WD + (size_t)(r - 1024) * Kdim;
    else if (r < 2064) src = Wb + (size_t)(r - 2048) * Kdim;
    else if (r < 2080) src = Wc + (size_t)(r - 2064) * Kdim;

    float4 v = make_float4(0.f, 0.f, 0.f, 0.f);
    if (src) v = *(const float4*)(src + k);

    uint2 pk;
    pk.x = (unsigned)f2bf(v.x) | ((unsigned)f2bf(v.y) << 16);
    pk.y = (unsigned)f2bf(v.z) | ((unsigned)f2bf(v.w) << 16);
    *(uint2*)&wcat[(size_t)r * Kdim + k] = pk;

    if (threadIdx.x == 0) {
        float b = 0.f;
        if      (r < 1024) b = bd[r];
        else if (r < 2048) b = bD[r - 1024];
        else if (r < 2064) b = bb[r - 2048];
        else if (r < 2080) b = bc[r - 2064];
        bcat[r] = b;
    }
}

// ---------------- kernel 1: layernorm -> h (bf16) ----------------
__global__ __launch_bounds__(256) void ln_kernel(
    const float* __restrict__ x, const float* __restrict__ w,
    const float* __restrict__ b, unsigned short* __restrict__ hb)
{
    __shared__ float s1[256];
    __shared__ float s2[256];
    const size_t row = blockIdx.x;
    const int t = threadIdx.x;
    const float* xr = x + row * Kdim;

    float4 v = *(const float4*)&xr[t * 4];
    float s = v.x + v.y + v.z + v.w;
    float q = v.x*v.x + v.y*v.y + v.z*v.z + v.w*v.w;
    s1[t] = s; s2[t] = q;
    __syncthreads();
    for (int o = 128; o > 0; o >>= 1) {
        if (t < o) { s1[t] += s1[t + o]; s2[t] += s2[t + o]; }
        __syncthreads();
    }
    float mean = s1[0] * (1.f / 1024.f);
    float var  = s2[0] * (1.f / 1024.f) - mean * mean;
    float rstd = rsqrtf(var + 1e-5f);

    float4 wv = *(const float4*)&w[t * 4];
    float4 bv = *(const float4*)&b[t * 4];
    float h0 = (v.x - mean) * rstd * wv.x + bv.x;
    float h1 = (v.y - mean) * rstd * wv.y + bv.y;
    float h2 = (v.z - mean) * rstd * wv.z + bv.z;
    float h3 = (v.w - mean) * rstd * wv.w + bv.w;

    uint2 pk;
    pk.x = (unsigned)f2bf(h0) | ((unsigned)f2bf(h1) << 16);
    pk.y = (unsigned)f2bf(h2) | ((unsigned)f2bf(h3) << 16);
    *(uint2*)&hb[row * Kdim + t * 4] = pk;
}

// ---------------- kernel 2: WMMA GEMM, TDM-staged + double buffered ----------------
__global__ __launch_bounds__(256) void gemm_kernel(
    const unsigned short* __restrict__ hb,    // (16384,1024) bf16
    const unsigned short* __restrict__ wcat,  // (2176,1024) bf16
    const float* __restrict__ bcat,           // (2176)
    const float* __restrict__ x,              // (16384,1024)
    float* __restrict__ delta,                // (16384,1024)
    float* __restrict__ outp,                 // (16384,1024)
    float* __restrict__ Bcb,                  // (16384,16)
    float* __restrict__ Ccb)                  // (16384,16)
{
    extern __shared__ char smem[];            // SMEM_BYTES, dynamic => LDS offset 0 base
    unsigned short* lds = (unsigned short*)smem;

    const int tid  = threadIdx.x;
    const int lane = tid & 31;
    const int wv   = tid >> 5;      // 0..7
    const int wm   = wv >> 2;       // 0..1 : M offset 64*wm
    const int wn   = wv & 3;        // 0..3 : N offset 32*wn
    const int m0   = blockIdx.x * BM;
    const int n0   = blockIdx.y * BN;

    v8f acc[4][2];
#pragma unroll
    for (int i = 0; i < 4; ++i)
#pragma unroll
        for (int j = 0; j < 2; ++j)
            acc[i][j] = (v8f){0.f,0.f,0.f,0.f,0.f,0.f,0.f,0.f};

    const int rsel = lane & 15;
    const int kb   = (lane >> 4) * 8;

#if defined(USE_TDM)
    const bool issuer = (wv == 0);  // one wave issues the DMA (TDM ignores EXEC)
    if (issuer) {
        tdm_load_tile(0,                  &hb[(size_t)m0 * Kdim]);
        tdm_load_tile(BM * BKP * 2,       &wcat[(size_t)n0 * Kdim]);
    }
    for (int k0 = 0, ks = 0; k0 < Kdim; k0 += BK, ++ks) {
        const int buf = ks & 1;
        if (k0 + BK < Kdim) {
            if (issuer) {
                unsigned nb = (unsigned)((buf ^ 1) * STAGE_BYTES);
                tdm_load_tile(nb,               &hb[(size_t)m0 * Kdim + k0 + BK]);
                tdm_load_tile(nb + BM * BKP * 2, &wcat[(size_t)n0 * Kdim + k0 + BK]);
                __builtin_amdgcn_s_wait_tensorcnt(2);   // stage k complete (in-order)
            }
        } else {
            if (issuer) __builtin_amdgcn_s_wait_tensorcnt(0);
        }
        __syncthreads();                                // stage k visible to all waves

        const unsigned short* ldsA = lds + buf * STAGE_USH;
        const unsigned short* ldsB = ldsA + BM * BKP;
#else
    for (int k0 = 0; k0 < Kdim; k0 += BK) {
        unsigned short* ldsA = lds;
        unsigned short* ldsB = lds + BM * BKP;
        // staged copy: issue all global loads, then all LDS stores
        uint4 ra[2], rb[2];
#pragma unroll
        for (int i = 0; i < 2; ++i) {
            int c   = tid + i * 256;
            int r   = c >> 2;
            int col = (c & 3) * 8;
            ra[i] = *(const uint4*)&hb[(size_t)(m0 + r) * Kdim + k0 + col];
            rb[i] = *(const uint4*)&wcat[(size_t)(n0 + r) * Kdim + k0 + col];
        }
#pragma unroll
        for (int i = 0; i < 2; ++i) {
            int c   = tid + i * 256;
            int r   = c >> 2;
            int col = (c & 3) * 8;
            *(uint4*)&ldsA[r * BKP + col] = ra[i];
            *(uint4*)&ldsB[r * BKP + col] = rb[i];
        }
        __syncthreads();
#endif
        v16bf afrag[4], bfrag[2];
#pragma unroll
        for (int i = 0; i < 4; ++i) {
            const unsigned short* p = &ldsA[(wm * 64 + i * 16 + rsel) * BKP + kb];
            v8bf lo = *(const v8bf*)p;
            v8bf hi = *(const v8bf*)(p + 16);
            afrag[i] = __builtin_shufflevector(lo, hi,
                0,1,2,3,4,5,6,7,8,9,10,11,12,13,14,15);
        }
#pragma unroll
        for (int j = 0; j < 2; ++j) {
            const unsigned short* p = &ldsB[(wn * 32 + j * 16 + rsel) * BKP + kb];
            v8bf lo = *(const v8bf*)p;
            v8bf hi = *(const v8bf*)(p + 16);
            bfrag[j] = __builtin_shufflevector(lo, hi,
                0,1,2,3,4,5,6,7,8,9,10,11,12,13,14,15);
        }
#pragma unroll
        for (int i = 0; i < 4; ++i)
#pragma unroll
            for (int j = 0; j < 2; ++j)
                acc[i][j] = __builtin_amdgcn_wmma_f32_16x16x32_bf16(
                    false, afrag[i], false, bfrag[j],
                    (short)0, acc[i][j], false, false);
        __syncthreads();
    }

    // epilogue: C layout — VGPR v, lanes 0-15 -> (M=v, N=lane); lanes 16-31 -> (M=v+8, N=lane-16)
    const int nrel  = lane & 15;
    const int mrel8 = (lane >> 4) * 8;
#pragma unroll
    for (int i = 0; i < 4; ++i) {
#pragma unroll
        for (int j = 0; j < 2; ++j) {
            int ncol = n0 + wn * 32 + j * 16 + nrel;
            float bias = bcat[ncol];
#pragma unroll
            for (int v = 0; v < 8; ++v) {
                size_t mrow = (size_t)(m0 + wm * 64 + i * 16 + mrel8 + v);
                float val = acc[i][j][v] + bias;
                if (ncol < 1024) {
                    float sp = (val > 20.f) ? val : log1pf(__expf(val));
                    delta[mrow * 1024 + ncol] = sp;               // softplus(h@Wd^T + bd)
                } else if (ncol < 2048) {
                    int nn = ncol - 1024;
                    outp[mrow * 1024 + nn] = x[mrow * 1024 + nn] + val;  // x + h@WD^T + bD
                } else if (ncol < 2064) {
                    Bcb[mrow * 16 + (ncol - 2048)] = val;          // h@Wb^T + bb
                } else if (ncol < 2080) {
                    Ccb[mrow * 16 + (ncol - 2064)] = val;          // h@Wc^T + bc
                }
            }
        }
    }
}

// ---------------- kernel 3: chunked scan, pass A (local scans, zero init) ----------------
__global__ __launch_bounds__(256) void scanA_kernel(
    const float* __restrict__ delta, const unsigned short* __restrict__ hb,
    const float* __restrict__ Bcb,   const float* __restrict__ logA,
    float* __restrict__ Sloc, float* __restrict__ sumdtb)
{
    int bid  = blockIdx.x;            // 512 = B * NCH * (D/256)
    int dblk = bid & 3;
    int c    = (bid >> 2) & 31;
    int b    = bid >> 7;
    int d    = dblk * 256 + threadIdx.x;

    float a[16];
    {
        const float* ap = &logA[(size_t)d * 16];
#pragma unroll
        for (int n = 0; n < 16; ++n) a[n] = -__expf(ap[n]);
    }
    float st[16];
#pragma unroll
    for (int n = 0; n < 16; ++n) st[n] = 0.f;
    float sdt = 0.f;

    const size_t base = (size_t)b * 4096 + (size_t)c * CHUNK;
    for (int t = 0; t < CHUNK; ++t) {
        size_t m = base + t;
        float dt = delta[m * 1024 + d];
        float h  = bf2f(hb[m * 1024 + d]);
        float dth = dt * h;
        sdt += dt;
        const float* Bp = &Bcb[m * 16];
#pragma unroll
        for (int n = 0; n < 16; ++n)
            st[n] = __expf(dt * a[n]) * st[n] + dth * Bp[n];
    }
    size_t cb = (size_t)c * 4 + b;
    size_t o  = (cb * 1024 + d) * 16;
#pragma unroll
    for (int n = 0; n < 16; ++n) Sloc[o + n] = st[n];
    sumdtb[cb * 1024 + d] = sdt;
}

// ---------------- kernel 4: sequential chunk combine (prefix over linear ops) ----------------
__global__ __launch_bounds__(256) void scan_combine_kernel(
    const float* __restrict__ logA, const float* __restrict__ sumdtb,
    const float* __restrict__ Sloc, float* __restrict__ Sinit)
{
    int g = blockIdx.x * 256 + threadIdx.x;   // 65536 = B*D*N
    int n = g & 15;
    int d = (g >> 4) & 1023;
    int b = g >> 14;
    float a = -__expf(logA[(size_t)d * 16 + n]);
    float init = 0.f;
    for (int c = 0; c < NCH; ++c) {
        size_t cb = (size_t)c * 4 + b;
        float P = __expf(a * sumdtb[cb * 1024 + d]);   // prod of exp(dt*A) over chunk
        size_t idx = (cb * 1024 + d) * 16 + n;
        Sinit[idx] = init;
        init = P * init + Sloc[idx];
    }
}

// ---------------- kernel 5: pass C (correct init, produce y, fused final add) ----------------
__global__ __launch_bounds__(256) void scanC_kernel(
    const float* __restrict__ delta, const unsigned short* __restrict__ hb,
    const float* __restrict__ Bcb,   const float* __restrict__ Ccb,
    const float* __restrict__ logA,  const float* __restrict__ Sinit,
    float* __restrict__ outp)
{
    int bid  = blockIdx.x;
    int dblk = bid & 3;
    int c    = (bid >> 2) & 31;
    int b    = bid >> 7;
    int d    = dblk * 256 + threadIdx.x;

    float a[16];
    {
        const float* ap = &logA[(size_t)d * 16];
#pragma unroll
        for (int n = 0; n < 16; ++n) a[n] = -__expf(ap[n]);
    }
    float st[16];
    {
        size_t cb = (size_t)c * 4 + b;
        size_t o  = (cb * 1024 + d) * 16;
#pragma unroll
        for (int n = 0; n < 16; ++n) st[n] = Sinit[o + n];
    }
    const size_t base = (size_t)b * 4096 + (size_t)c * CHUNK;
    for (int t = 0; t < CHUNK; ++t) {
        size_t m = base + t;
        float dt = delta[m * 1024 + d];
        float h  = bf2f(hb[m * 1024 + d]);
        float dth = dt * h;
        const float* Bp = &Bcb[m * 16];
        const float* Cp = &Ccb[m * 16];
        float y = 0.f;
#pragma unroll
        for (int n = 0; n < 16; ++n) {
            st[n] = __expf(dt * a[n]) * st[n] + dth * Bp[n];
            y += st[n] * Cp[n];
        }
        outp[m * 1024 + d] += y;    // out already holds x + h@WD^T + bD
    }
}

// ---------------- launcher ----------------
extern "C" void kernel_launch(void* const* d_in, const int* in_sizes, int n_in,
                              void* d_out, int out_size, void* d_ws, size_t ws_size,
                              hipStream_t stream)
{
    const float* x    = (const float*)d_in[0];
    const float* logA = (const float*)d_in[1];
    const float* Wd   = (const float*)d_in[2];
    const float* bd   = (const float*)d_in[3];
    const float* Wb   = (const float*)d_in[4];
    const float* bb   = (const float*)d_in[5];
    const float* Wc   = (const float*)d_in[6];
    const float* bc   = (const float*)d_in[7];
    const float* WD   = (const float*)d_in[8];
    const float* bD   = (const float*)d_in[9];
    const float* lnw  = (const float*)d_in[10];
    const float* lnb  = (const float*)d_in[11];
    float* outp = (float*)d_out;

    char* ws = (char*)d_ws;
    size_t off = 0;
    auto alloc = [&](size_t bytes) -> void* {
        void* p = ws + off;
        off = (off + bytes + 255) & ~(size_t)255;
        return p;
    };
    unsigned short* hb    = (unsigned short*)alloc((size_t)Mrows * Kdim * 2);
    unsigned short* wcat  = (unsigned short*)alloc((size_t)Ncat * Kdim * 2);
    float*          bcat  = (float*)alloc((size_t)Ncat * 4);
    float*          delta = (float*)alloc((size_t)Mrows * Kdim * 4);
    float*          Bcb   = (float*)alloc((size_t)Mrows * 16 * 4);
    float*          Ccb   = (float*)alloc((size_t)Mrows * 16 * 4);
    float*          Sloc  = (float*)alloc((size_t)NCH * 4 * 1024 * 16 * 4);
    float*          Sinit = (float*)alloc((size_t)NCH * 4 * 1024 * 16 * 4);
    float*          sdt   = (float*)alloc((size_t)NCH * 4 * 1024 * 4);

    pack_kernel<<<Ncat, 256, 0, stream>>>(Wd, bd, Wb, bb, Wc, bc, WD, bD, wcat, bcat);
    ln_kernel<<<Mrows, 256, 0, stream>>>(x, lnw, lnb, hb);
    gemm_kernel<<<dim3(Mrows / BM, Ncat / BN), 256, SMEM_BYTES, stream>>>(
        hb, wcat, bcat, x, delta, outp, Bcb, Ccb);
    scanA_kernel<<<512, 256, 0, stream>>>(delta, hb, Bcb, logA, Sloc, sdt);
    scan_combine_kernel<<<256, 256, 0, stream>>>(logA, sdt, Sloc, Sinit);
    scanC_kernel<<<512, 256, 0, stream>>>(delta, hb, Bcb, Ccb, logA, Sinit, outp);
}
